// AttentionBlock_85684597555643
// MI455X (gfx1250) — matrix operand order repaired and checked by reading, not compiled, not run on hardware
//
#include <hip/hip_runtime.h>

typedef __bf16 bf16_t;
typedef __bf16 v16bf __attribute__((ext_vector_type(16)));
typedef float  v8f   __attribute__((ext_vector_type(8)));

#define B_   8
#define C_   256
#define C8_  32
#define N_   4096
#define QB_  64
#define KT_  32
#define KS_  40   // 32 + 8 pad bf16 -> 80B rows: 16B aligned, low-conflict (20 dwords)

// DPP16 controls for 16-lane butterfly reductions (wave32: two independent rows)
#define DPP_XOR1  0xB1   // quad_perm [1,0,3,2]
#define DPP_XOR2  0x4E   // quad_perm [2,3,0,1]
#define DPP_HMIRR 0x141  // row_half_mirror (combines the two quads of each 8-group)
#define DPP_MIRR  0x140  // row_mirror      (combines the two 8-groups of each row)

#define DPP_MAXRED(x, ctrl) \
    x = fmaxf(x, __builtin_bit_cast(float, __builtin_amdgcn_mov_dpp( \
        __builtin_bit_cast(int, x), (ctrl), 0xF, 0xF, true)))
#define DPP_SUMRED(x, ctrl) \
    x = x + __builtin_bit_cast(float, __builtin_amdgcn_mov_dpp( \
        __builtin_bit_cast(int, x), (ctrl), 0xF, 0xF, true))

union F16 { uint4 q[2]; v16bf v; };   // 16 bf16 fragment as two b128 halves

static __device__ __forceinline__ bf16_t f2bf(float f) {
    unsigned u = __builtin_bit_cast(unsigned, f);
    u += 0x7FFFu + ((u >> 16) & 1u);          // round-to-nearest-even
    unsigned short h = (unsigned short)(u >> 16);
    return __builtin_bit_cast(bf16_t, h);
}

// ---------------- projections: q/k/v = W*x + b (channel GEMMs) ----------------
__global__ __launch_bounds__(256) void proj_kernel(
    const float* __restrict__ x,
    const float* __restrict__ Wq, const float* __restrict__ bq,
    const float* __restrict__ Wk, const float* __restrict__ bk,
    const float* __restrict__ Wv, const float* __restrict__ bv,
    bf16_t* __restrict__ qg, bf16_t* __restrict__ kg, bf16_t* __restrict__ vg)
{
    const int b  = blockIdx.z;
    const int r0 = blockIdx.y * 4;                 // 4 rows of stacked [320,256] weights
    const int n  = blockIdx.x * 256 + threadIdx.x; // spatial position

    const float* W; const float* bias;
    if (r0 < 32)      { W = Wq + (size_t)r0 * C_;       bias = bq + r0;       }
    else if (r0 < 64) { W = Wk + (size_t)(r0-32) * C_;  bias = bk + (r0-32);  }
    else              { W = Wv + (size_t)(r0-64) * C_;  bias = bv + (r0-64);  }

    float a0 = bias[0], a1 = bias[1], a2 = bias[2], a3 = bias[3];
    const float* xb = x + (size_t)b * C_ * N_ + n;
    #pragma unroll 4
    for (int c = 0; c < C_; ++c) {
        float xv = xb[(size_t)c * N_];             // coalesced across threads
        a0 = fmaf(W[c],        xv, a0);
        a1 = fmaf(W[C_   + c], xv, a1);
        a2 = fmaf(W[2*C_ + c], xv, a2);
        a3 = fmaf(W[3*C_ + c], xv, a3);
    }

    if (r0 < 32) {                                  // q: [B][N][32] (A-fragment friendly)
        bf16_t* q = qg + ((size_t)b * N_ + n) * C8_ + r0;
        q[0] = f2bf(a0); q[1] = f2bf(a1); q[2] = f2bf(a2); q[3] = f2bf(a3);
    } else if (r0 < 64) {                           // k: [B][N][32] (key-major, B-tile friendly)
        bf16_t* k = kg + ((size_t)b * N_ + n) * C8_ + (r0-32);
        k[0] = f2bf(a0); k[1] = f2bf(a1); k[2] = f2bf(a2); k[3] = f2bf(a3);
    } else {                                        // v: [B][256][N]
        bf16_t* v = vg + ((size_t)b * C_ + (r0-64)) * N_ + n;
        v[0] = f2bf(a0); v[N_] = f2bf(a1); v[2*N_] = f2bf(a2); v[3*N_] = f2bf(a3);
    }
}

// ---------------- flash attention + fused residual epilogue ----------------
__global__ __launch_bounds__(256) void attn_kernel(
    const bf16_t* __restrict__ qg, const bf16_t* __restrict__ kg,
    const bf16_t* __restrict__ vg, const float* __restrict__ x,
    const float* __restrict__ gamma, float* __restrict__ out)
{
    __shared__ __align__(16) bf16_t lds_kt[2][KT_][KS_];  //  5 KB : K tiles [key][dim], double-buffered
    __shared__ __align__(16) bf16_t lds_v[2][C_][KS_];    // 40 KB : V tiles [ch][key], double-buffered
    __shared__ __align__(16) bf16_t lds_p[8][16][KS_];    // 10 KB : per-wave P transpose patch

    const int b    = blockIdx.y;
    const int qblk = blockIdx.x;
    const int tid  = threadIdx.x;
    const int wave = tid >> 5;
    const int lane = tid & 31;
    const int hi   = lane >> 4;     // half-wave group (rows 0-7 vs 8-15 in C/D layout)
    const int l15  = lane & 15;
    const int qt   = wave & 3;      // 16-query sub-tile within the 64-query block
    const int half = wave >> 2;     // 128-channel half
    const int q0   = qblk * QB_ + qt * 16;

    // ---- per-lane async-DMA assignments (5 ops/wave/tile: 1 K b64 + 4 V b128) ----
    const int kj    = wave * 4 + (lane >> 3);            // key row (wave covers 4 keys)
    const int kd0   = (lane & 7) * 4;                    // dim start (8B chunk)
    const unsigned kvoff = (unsigned)(kj * (C8_ * 2) + kd0 * 2);
    const int vchunk = lane & 3;                         // 16B chunk within 64B key-row

    auto issue_async_tile = [&](int bufsel, int m0) {
        unsigned long long kbase = (unsigned long long)(size_t)(kg + ((size_t)b * N_ + m0) * C8_);
        unsigned klds = (unsigned)(size_t)&lds_kt[bufsel][kj][kd0];
        asm volatile("global_load_async_to_lds_b64 %0, %1, %2"
                     :: "v"(klds), "v"(kvoff), "s"(kbase) : "memory");
        unsigned long long vbase = (unsigned long long)(size_t)(vg + (size_t)b * C_ * N_ + m0);
        #pragma unroll
        for (int i = 0; i < 4; ++i) {
            int ch = wave * 32 + i * 8 + (lane >> 2);    // wave covers 32 channels
            unsigned vlds  = (unsigned)(size_t)&lds_v[bufsel][ch][vchunk * 8];
            unsigned vvoff = (unsigned)(ch * (N_ * 2) + vchunk * 16);
            asm volatile("global_load_async_to_lds_b128 %0, %1, %2"
                         :: "v"(vlds), "v"(vvoff), "s"(vbase) : "memory");
        }
    };

    // ---- Q A-fragment: per-lane K runs [hi*8, +8) and [16+hi*8, +8) -> 2 x b128 ----
    F16 aq;
    {
        const bf16_t* qrow = qg + ((size_t)b * N_ + (q0 + l15)) * C8_;
        aq.q[0] = *(const uint4*)(qrow + hi * 8);
        aq.q[1] = *(const uint4*)(qrow + 16 + hi * 8);
    }

    v8f o[8];                       // 16q x 128ch fp32 accumulators
    #pragma unroll
    for (int t = 0; t < 8; ++t) {
        #pragma unroll
        for (int j = 0; j < 8; ++j) o[t][j] = 0.0f;
    }
    v8f czero;
    #pragma unroll
    for (int j = 0; j < 8; ++j) czero[j] = 0.0f;

    float mrun[8], lrun[8];
    #pragma unroll
    for (int j = 0; j < 8; ++j) { mrun[j] = -3.0e38f; lrun[j] = 0.0f; }

    issue_async_tile(0, 0);         // prologue: tile 0 -> buffer 0

    for (int ktile = 0; ktile < N_ / KT_; ++ktile) {
        const int buf = ktile & 1;
        // overlap: DMA tile t+1 into the other buffer while computing tile t
        // (last iteration issues a harmless wrapped tile; s_endpgm drains it)
        issue_async_tile(buf ^ 1, ((ktile + 1) * KT_) & (N_ - 1));
        asm volatile("s_wait_asynccnt 0x5" ::: "memory");   // tile t's 5 ops retired (in-order)
        __syncthreads();

        // ---- S = Q x K  (two 16-key WMMAs, K-dim = 32 = full C8) ----
        F16 bk0, bk1;
        {
            const bf16_t* r0p = &lds_kt[buf][l15][0];      // B-layout: lane dim-runs [hi*8,+8) and [16+hi*8,+8)
            const bf16_t* r1p = &lds_kt[buf][16 + l15][0];
            bk0.q[0] = ((const uint4*)r0p)[hi]; bk0.q[1] = ((const uint4*)r0p)[2 + hi];
            bk1.q[0] = ((const uint4*)r1p)[hi]; bk1.q[1] = ((const uint4*)r1p)[2 + hi];
        }
        v8f s0 = __builtin_amdgcn_wmma_f32_16x16x32_bf16(false, aq.v, false, bk0.v, (short)0, czero, false, false);
        v8f s1 = __builtin_amdgcn_wmma_f32_16x16x32_bf16(false, aq.v, false, bk1.v, (short)0, czero, false, false);

        // ---- online softmax over this 32-key tile (DPP16 butterfly, no LDS) ----
        float sc[8];
        #pragma unroll
        for (int j = 0; j < 8; ++j) {
            float tmax = fmaxf(s0[j], s1[j]);
            DPP_MAXRED(tmax, DPP_XOR1);
            DPP_MAXRED(tmax, DPP_XOR2);
            DPP_MAXRED(tmax, DPP_HMIRR);
            DPP_MAXRED(tmax, DPP_MIRR);
            float mnew  = fmaxf(mrun[j], tmax);
            float scale = __expf(mrun[j] - mnew);
            float p0 = __expf(s0[j] - mnew);
            float p1 = __expf(s1[j] - mnew);
            float rs = p0 + p1;
            DPP_SUMRED(rs, DPP_XOR1);
            DPP_SUMRED(rs, DPP_XOR2);
            DPP_SUMRED(rs, DPP_HMIRR);
            DPP_SUMRED(rs, DPP_MIRR);
            mrun[j] = mnew;
            lrun[j] = lrun[j] * scale + rs;
            sc[j]   = scale;
            s0[j] = p0; s1[j] = p1;
        }
        #pragma unroll
        for (int t = 0; t < 8; ++t) {
            #pragma unroll
            for (int j = 0; j < 8; ++j) o[t][j] *= sc[j];
        }

        // ---- P: accumulator layout -> A layout via per-wave private LDS patch ----
        #pragma unroll
        for (int j = 0; j < 8; ++j) {
            lds_p[wave][hi*8 + j][l15]      = f2bf(s0[j]);
            lds_p[wave][hi*8 + j][16 + l15] = f2bf(s1[j]);
        }
        asm volatile("s_wait_dscnt 0" ::: "memory");   // in-wave DS RAW, no block barrier needed
        F16 ap;
        {
            const bf16_t* prow = &lds_p[wave][l15][0];
            ap.q[0] = *(const uint4*)(prow + hi * 8);
            ap.q[1] = *(const uint4*)(prow + 16 + hi * 8);
        }

        // ---- O += P x V^T  (8 channel sub-tiles of 16) ----
        #pragma unroll
        for (int t = 0; t < 8; ++t) {
            F16 bv;
            int ch = half*128 + t*16 + l15;
            const bf16_t* vrow = &lds_v[buf][ch][0];
            bv.q[0] = ((const uint4*)vrow)[hi];
            bv.q[1] = ((const uint4*)vrow)[2 + hi];
            o[t] = __builtin_amdgcn_wmma_f32_16x16x32_bf16(false, ap.v, false, bv.v, (short)0, o[t], false, false);
        }
        __syncthreads();            // protect the buffer refilled next iteration
    }

    // ---- epilogue: out = gamma * (O / l) + x ----
    const float g = gamma[0];
    float linv[8];
    #pragma unroll
    for (int j = 0; j < 8; ++j) linv[j] = 1.0f / lrun[j];
    #pragma unroll
    for (int t = 0; t < 8; ++t) {
        int ch = half*128 + t*16 + l15;
        #pragma unroll
        for (int j = 0; j < 8; ++j) {
            int qrow = q0 + hi*8 + j;
            size_t idx = ((size_t)b * C_ + ch) * N_ + qrow;
            out[idx] = g * (o[t][j] * linv[j]) + x[idx];
        }
    }
}

extern "C" void kernel_launch(void* const* d_in, const int* in_sizes, int n_in,
                              void* d_out, int out_size, void* d_ws, size_t ws_size,
                              hipStream_t stream) {
    (void)in_sizes; (void)n_in; (void)out_size; (void)ws_size;
    const float* x     = (const float*)d_in[0];
    const float* Wq    = (const float*)d_in[1];
    const float* bq    = (const float*)d_in[2];
    const float* Wk    = (const float*)d_in[3];
    const float* bk    = (const float*)d_in[4];
    const float* Wv    = (const float*)d_in[5];
    const float* bv    = (const float*)d_in[6];
    const float* gamma = (const float*)d_in[7];

    bf16_t* qg = (bf16_t*)d_ws;                      //  2 MB  [B][N][32]
    bf16_t* kg = qg + (size_t)B_ * N_ * C8_;         //  2 MB  [B][N][32]
    bf16_t* vg = kg + (size_t)B_ * N_ * C8_;         // 16 MB  [B][256][N]

    proj_kernel<<<dim3(16, 80, 8), 256, 0, stream>>>(x, Wq, bq, Wk, bk, Wv, bv, qg, kg, vg);
    attn_kernel<<<dim3(N_ / QB_, B_), 256, 0, stream>>>(qg, kg, vg, x, gamma, (float*)d_out);
}
